// ConFormer_14164802142501
// MI455X (gfx1250) — compile-verified, matrix operand order checked by reading, not run on hardware
//
#include <hip/hip_runtime.h>
#include <hip/hip_bf16.h>
#include <math.h>

typedef __attribute__((ext_vector_type(16))) _Float16 v16h;
typedef __attribute__((ext_vector_type(8)))  float    v8f;

#define B_ 32
#define S_ 12
#define N_ 1024
#define D_ 64
#define H_ 64
#define K_ 16
#define L_ 16
#define G_ 192           // 3*H
#define R_ 8192          // B_*K_*L_ rows of XW
#define ALPHA_ 0.15f

// f16 weight block layout inside d_ws (element offsets in _Float16 units)
#define W16_SPW1 0            // 64*64   = 4096
#define W16_SCW1 4096         // 64*128  = 8192
#define W16_WIHF 12288        // 192*64  = 12288
#define W16_WIHB 24576
#define W16_WHHF 36864
#define W16_WHHB 49152
#define W16_TOTAL 61440

// ---------------- WMMA fragment helpers (16x16x32 f16 -> f32) ----------------
// A (16x32, row-major, row = lane&15): lanes 0-15 hold K {0..7,16..23},
// lanes 16-31 hold K {8..15,24..31} (ISA 7.12.2 16-bit A layout).
// B = W^T gathered from row-major W (N x K): same gather with row = n.
__device__ __forceinline__ v16h frag_gather_f32(const float* rowp, int lane) {
  const int h = (lane >> 4) & 1;
  v16h a;
  #pragma unroll
  for (int i = 0; i < 8; ++i) {
    a[i]     = (_Float16)rowp[h * 8 + i];
    a[8 + i] = (_Float16)rowp[16 + h * 8 + i];
  }
  return a;
}

// f16 source: two contiguous 16-byte runs per lane -> vectorizable b128 loads.
__device__ __forceinline__ v16h frag_gather_f16(const _Float16* rowp, int lane) {
  const int h = (lane >> 4) & 1;
  const _Float16* p0 = rowp + h * 8;
  const _Float16* p1 = rowp + 16 + h * 8;
  v16h a;
  #pragma unroll
  for (int i = 0; i < 8; ++i) { a[i] = p0[i]; a[8 + i] = p1[i]; }
  return a;
}

// C/D 16x16 f32: lane column n = lane&15, row = i + 8*(lane>>4).
__device__ __forceinline__ void store_tile(float* C, int ldc, v8f d, int lane) {
  const int h = (lane >> 4) & 1, n = lane & 15;
  #pragma unroll
  for (int i = 0; i < 8; ++i) C[(size_t)(i + 8 * h) * ldc + n] = d[i];
}

__device__ __forceinline__ float sigmf(float x) { return 1.f / (1.f + expf(-x)); }

__device__ __forceinline__ float rnd(unsigned a, unsigned b, unsigned c) {
  unsigned x = a * 0x9E3779B9u ^ b * 0x85EBCA6Bu ^ c * 0xC2B2AE35u;
  x ^= x >> 16; x *= 0x7FEB352Du; x ^= x >> 15; x *= 0x846CA68Bu; x ^= x >> 16;
  return (float)(x >> 8) * (1.0f / 16777216.0f);
}

// ---------------- 0. one-shot weight conversion to f16 ----------------
__global__ void prep_kernel(const float* __restrict__ sp_w1, const float* __restrict__ sc_w1,
                            const float* __restrict__ wih_f, const float* __restrict__ wih_b,
                            const float* __restrict__ whh_f, const float* __restrict__ whh_b,
                            _Float16* __restrict__ o) {
  const int i = blockIdx.x * 256 + threadIdx.x;   // 61440 total
  if (i < W16_SCW1)            o[i] = (_Float16)sp_w1[i - W16_SPW1];
  else if (i < W16_WIHF)       o[i] = (_Float16)sc_w1[i - W16_SCW1];
  else if (i < W16_WIHB)       o[i] = (_Float16)wih_f[i - W16_WIHF];
  else if (i < W16_WHHF)       o[i] = (_Float16)wih_b[i - W16_WIHB];
  else if (i < W16_WHHB)       o[i] = (_Float16)whh_f[i - W16_WHHF];
  else if (i < W16_TOTAL)      o[i] = (_Float16)whh_b[i - W16_WHHB];
}

// ---------------- 1. feature_change ----------------
__global__ void fc_kernel(const float* __restrict__ nf, float* __restrict__ fc) {
  const int bn = blockIdx.x;            // B_*N_ blocks
  const int b = bn >> 10, n = bn & 1023;
  const int d = threadIdx.x;            // 64 threads
  const float* base = nf + (((size_t)b * S_) * N_ + n) * (size_t)D_ + d;
  float prev = base[0], s = 0.f;
  #pragma unroll
  for (int t = 1; t < S_; ++t) {
    float c = base[(size_t)t * N_ * D_];
    s += fabsf(c - prev); prev = c;
  }
  __shared__ float red[64];
  red[d] = s; __syncthreads();
  if (d == 0) {
    float tot = 0.f;
    for (int i = 0; i < 64; ++i) tot += red[i];
    fc[bn] = tot / (float)((S_ - 1) * D_);
  }
}

// ---------------- 2. source scores (WMMA MLP) ----------------
__global__ void src_score_kernel(const float* __restrict__ nf,
                                 const _Float16* __restrict__ w16,
                                 const float* __restrict__ b1,
                                 const float* __restrict__ w2, const float* __restrict__ b2,
                                 const float* __restrict__ fc, float* __restrict__ score) {
  __shared__ float rowsum[8][16];
  const int wid = threadIdx.x >> 5, lane = threadIdx.x & 31;
  const int tile = blockIdx.x * 8 + wid;      // 2048 tiles over 32768 rows
  const int row0 = tile * 16;
  if (lane < 16) rowsum[wid][lane] = 0.f;
  __syncthreads();
  const int m = lane & 15, h = (lane >> 4) & 1;
  const int row = row0 + m, b = row >> 10, n = row & 1023;
  const float* arow = nf + (((size_t)b * S_ + (S_ - 1)) * N_ + n) * (size_t)D_;
  const v16h a0 = frag_gather_f32(arow, lane);        // single use: convert inline
  const v16h a1 = frag_gather_f32(arow + 32, lane);
  const _Float16* w1h = w16 + W16_SPW1;
  #pragma unroll
  for (int j = 0; j < 4; ++j) {
    const _Float16* wrow = w1h + (size_t)(j * 16 + m) * 64;
    const v16h bf0 = frag_gather_f16(wrow, lane);
    const v16h bf1 = frag_gather_f16(wrow + 32, lane);
    v8f c = {};
    c = __builtin_amdgcn_wmma_f32_16x16x32_f16(false, a0, false, bf0, (short)0, c, false, false);
    c = __builtin_amdgcn_wmma_f32_16x16x32_f16(false, a1, false, bf1, (short)0, c, false, false);
    const int col = j * 16 + m;
    const float bb = b1[col], wv = w2[col];
    #pragma unroll
    for (int i = 0; i < 8; ++i) {
      float t = c[i] + bb; t = t > 0.f ? t : 0.f;
      atomicAdd(&rowsum[wid][i + 8 * h], t * wv);
    }
  }
  __syncthreads();
  if (lane < 16) {
    const int r = row0 + lane;
    score[r] = (rowsum[wid][lane] + b2[0]) * fc[r];
  }
}

// ---------------- 3. per-batch top-K ----------------
__global__ void topk_kernel(const float* __restrict__ score, int* __restrict__ srcidx) {
  const int b = blockIdx.x, t = threadIdx.x;  // 256 threads
  __shared__ float sv[1024];
  __shared__ float rv[256];
  __shared__ int   ri[256];
  for (int j = 0; j < 4; ++j) sv[t + 256 * j] = score[b * 1024 + t + 256 * j];
  __syncthreads();
  for (int it = 0; it < K_; ++it) {
    float bv = -3.4e38f; int bi = 0;
    for (int j = 0; j < 4; ++j) {
      const int n = t + 256 * j;
      if (sv[n] > bv) { bv = sv[n]; bi = n; }
    }
    rv[t] = bv; ri[t] = bi; __syncthreads();
    for (int s = 128; s > 0; s >>= 1) {
      if (t < s && rv[t + s] > rv[t]) { rv[t] = rv[t + s]; ri[t] = ri[t + s]; }
      __syncthreads();
    }
    if (t == 0) { srcidx[b * K_ + it] = ri[0]; sv[ri[0]] = -3.4e38f; }
    __syncthreads();
  }
}

// ---------------- 4. guided walks (one block per walk) ----------------
__global__ void walk_kernel(const float* __restrict__ nf, const float* __restrict__ adj,
                            const float* __restrict__ fc, const int* __restrict__ srcidx,
                            int* __restrict__ paths, float* __restrict__ out_paths) {
  const int bk = blockIdx.x, b = bk / K_, t = threadIdx.x;   // 256 threads
  __shared__ float p2[1024];
  __shared__ unsigned char vis[1024];
  __shared__ float red[256];
  __shared__ float lcur[64];
  __shared__ int cur_s, done_s;
  __shared__ float fcsum_s;

  const float* last = nf + (((size_t)b * S_ + (S_ - 1)) * N_) * (size_t)D_;
  const float* fcb = fc + b * 1024;
  float fs = 0.f;
  for (int j = 0; j < 4; ++j) fs += fcb[t + 256 * j];
  red[t] = fs; __syncthreads();
  for (int s = 128; s > 0; s >>= 1) { if (t < s) red[t] += red[t + s]; __syncthreads(); }
  const int source = srcidx[bk];
  for (int j = 0; j < 4; ++j) vis[t + 256 * j] = 0;
  if (t == 0) {
    fcsum_s = red[0] + 1e-8f;
    vis[source] = 1; cur_s = source; done_s = 0;
    paths[bk * L_] = source; out_paths[bk * L_] = (float)source;
  }
  __syncthreads();

  for (int step = 1; step < L_; ++step) {
    const int cur = cur_s;
    if (t < 64) lcur[t] = last[(size_t)cur * D_ + t];
    __syncthreads();
    float lmax = 0.f;
    for (int j = 0; j < 4; ++j) {
      const int n = t + 256 * j;
      const float* fn = last + (size_t)n * D_;
      float s2 = 0.f;
      #pragma unroll 8
      for (int d = 0; d < 64; ++d) { const float dv = fn[d] - lcur[d]; s2 += dv * dv; }
      const float dd = sqrtf(s2);
      p2[n] = dd;
      lmax = dd > lmax ? dd : lmax;
    }
    red[t] = lmax; __syncthreads();
    for (int s = 128; s > 0; s >>= 1) { if (t < s) red[t] = red[t] > red[t + s] ? red[t] : red[t + s]; __syncthreads(); }
    const float m = red[0];
    __syncthreads();
    const float* arow = adj + ((size_t)b * N_ + cur) * (size_t)N_;
    float psum = 0.f;
    for (int j = 0; j < 4; ++j) {
      const int n = t + 256 * j;
      const float g = m > 0.f ? p2[n] / m : 0.f;
      const float pr = arow[n] * (vis[n] ? 0.f : 1.f) * (1.f + fcb[n] / fcsum_s) * (1.f + g);
      p2[n] = pr;
      psum += pr;
    }
    red[t] = psum; __syncthreads();
    for (int s = 128; s > 0; s >>= 1) { if (t < s) red[t] += red[t + s]; __syncthreads(); }
    if (t == 0) {
      const float S = red[0];
      const float u1 = rnd((unsigned)bk, (unsigned)step, 1u);
      const float u2 = rnd((unsigned)bk, (unsigned)step, 2u);
      const int restart = u1 < ALPHA_;
      int nxt;
      if (restart || S <= 0.f) nxt = source;
      else {
        const float u = u2 * S; float c = 0.f; nxt = N_ - 1;
        for (int n = 0; n < N_; ++n) { c += p2[n]; if (c >= u) { nxt = n; break; } }
      }
      const int done2 = done_s | ((!restart) && (S <= 0.f));
      const int outn = done2 ? source : nxt;
      if (!done2) { cur_s = nxt; vis[nxt] = 1; }
      done_s = done2;
      paths[bk * L_ + step] = outn;
      out_paths[bk * L_ + step] = (float)outn;
      __builtin_prefetch(adj + ((size_t)b * N_ + cur_s) * (size_t)N_, 0, 1); // global_prefetch_b8
    }
    __syncthreads();
  }
}

// ---------------- 5. gather path features (write f16 directly) ----------------
__global__ void gather_kernel(const float* __restrict__ nf, const int* __restrict__ paths,
                              _Float16* __restrict__ x16) {
  const int idx = blockIdx.x * 256 + threadIdx.x;   // 512*16*64 elements
  const int d = idx & 63, l = (idx >> 6) & 15, bk = idx >> 10, b = bk / K_;
  const int node = paths[bk * L_ + l];
  x16[idx] = (_Float16)nf[(((size_t)b * S_ + (S_ - 1)) * N_ + node) * (size_t)D_ + d];
}

// ---------------- 6. GRU input projection, both directions (WMMA, f16 in) ----------------
__global__ void xw_kernel(const _Float16* __restrict__ x16, const _Float16* __restrict__ w16,
                          float* __restrict__ xw) {
  const int wid = threadIdx.x >> 5, lane = threadIdx.x & 31;
  const int task = blockIdx.x * 8 + wid;        // 12288 wave tasks
  const int dir = task / 6144, rem = task % 6144;
  const int mtile = rem / 12, ntile = rem % 12;
  const _Float16* W = w16 + (dir ? W16_WIHB : W16_WIHF);
  const int m = lane & 15;
  const int r = mtile * 16 + m, seq = r >> 4, tt = r & 15;
  const int te = dir ? (15 - tt) : tt;          // fold time reversal into gather
  const _Float16* arow = x16 + (size_t)(seq * 16 + te) * 64;
  const v16h a0 = frag_gather_f16(arow, lane);
  const v16h a1 = frag_gather_f16(arow + 32, lane);
  const _Float16* wrow = W + (size_t)(ntile * 16 + m) * 64;
  const v16h b0 = frag_gather_f16(wrow, lane);
  const v16h b1 = frag_gather_f16(wrow + 32, lane);
  v8f c = {};
  c = __builtin_amdgcn_wmma_f32_16x16x32_f16(false, a0, false, b0, (short)0, c, false, false);
  c = __builtin_amdgcn_wmma_f32_16x16x32_f16(false, a1, false, b1, (short)0, c, false, false);
  float* C = xw + (size_t)dir * R_ * G_ + (size_t)(mtile * 16) * G_ + ntile * 16;
  store_tile(C, G_, c, lane);
}

// ---------------- 7. GRU scan (h in registers + f16 shadow in LDS) ----------------
__global__ void gru_scan_kernel(const float* __restrict__ xw, const _Float16* __restrict__ w16,
                                const float* __restrict__ bih_f, const float* __restrict__ bhh_f,
                                const float* __restrict__ bih_b, const float* __restrict__ bhh_b,
                                float* __restrict__ pf) {
  const int blk = blockIdx.x;                   // 64 blocks
  const int dir = blk & 1, mtile = blk >> 1;
  const _Float16* whh = w16 + (dir ? W16_WHHB : W16_WHHF);
  const float* bih = dir ? bih_b : bih_f;
  const float* bhh = dir ? bhh_b : bhh_f;
  __shared__ _Float16 hh[16 * 64];              // f16 hidden state feeding WMMA A
  __shared__ float hw[16 * 192];
  const int tid = threadIdx.x;                  // 128 threads = 4 waves
  const int wid = tid >> 5, lane = tid & 31;
  for (int i = tid; i < 16 * 64; i += 128) hh[i] = (_Float16)0.f;
  __syncthreads();
  const int steps = dir ? 1 : 16;               // bwd output = gru(x_rev)[:,0]
  const float* xwd = xw + (size_t)dir * R_ * G_;
  const int m = tid >> 3, d0 = (tid & 7) * 8;
  const int mm = lane & 15;
  float hreg[8];                                // this thread's h[m][d0..d0+7]
  #pragma unroll
  for (int q = 0; q < 8; ++q) hreg[q] = 0.f;
  for (int t = 0; t < steps; ++t) {
    // hw = h @ whh^T : 12 N-tiles over 4 waves, K=64 in 2 WMMA steps
    const _Float16* arow = hh + mm * 64;
    const v16h a0 = frag_gather_f16(arow, lane);
    const v16h a1 = frag_gather_f16(arow + 32, lane);
    #pragma unroll
    for (int jj = 0; jj < 3; ++jj) {
      const int ntile = wid * 3 + jj;
      const _Float16* wrow = whh + (size_t)(ntile * 16 + mm) * 64;
      const v16h b0 = frag_gather_f16(wrow, lane);
      const v16h b1 = frag_gather_f16(wrow + 32, lane);
      v8f c = {};
      c = __builtin_amdgcn_wmma_f32_16x16x32_f16(false, a0, false, b0, (short)0, c, false, false);
      c = __builtin_amdgcn_wmma_f32_16x16x32_f16(false, a1, false, b1, (short)0, c, false, false);
      store_tile(hw + ntile * 16, 192, c, lane);
    }
    __syncthreads();
    const int seq = mtile * 16 + m;
    const float* xr = xwd + (size_t)(seq * 16 + t) * 192;
    #pragma unroll
    for (int q = 0; q < 8; ++q) {
      const int d = d0 + q;
      const float xg_r = xr[d]       + bih[d];
      const float xg_z = xr[64 + d]  + bih[64 + d];
      const float xg_n = xr[128 + d] + bih[128 + d];
      const float hg_r = hw[m * 192 + d]       + bhh[d];
      const float hg_z = hw[m * 192 + 64 + d]  + bhh[64 + d];
      const float hg_n = hw[m * 192 + 128 + d] + bhh[128 + d];
      const float rg = sigmf(xg_r + hg_r);
      const float zg = sigmf(xg_z + hg_z);
      const float ng = tanhf(xg_n + rg * hg_n);
      const float h2 = (1.f - zg) * ng + zg * hreg[q];
      hreg[q] = h2;
      hh[m * 64 + d] = (_Float16)h2;
      if (t == steps - 1) pf[(size_t)seq * 128 + dir * 64 + d] = h2;
    }
    __syncthreads();
  }
}

// ---------------- 8. path scoring MLP (WMMA) ----------------
__global__ void pathscore_kernel(const float* __restrict__ pf, const _Float16* __restrict__ w16,
                                 const float* __restrict__ b1,
                                 const float* __restrict__ w2, const float* __restrict__ b2,
                                 float* __restrict__ pscore) {
  __shared__ float rowsum[8][16];
  const int wid = threadIdx.x >> 5, lane = threadIdx.x & 31;
  const int tile = blockIdx.x * 8 + wid;  // 32 tiles over 512 rows
  const int row0 = tile * 16;
  if (lane < 16) rowsum[wid][lane] = 0.f;
  __syncthreads();
  const int m = lane & 15, h = (lane >> 4) & 1;
  const float* arow = pf + (size_t)(row0 + m) * 128;
  v16h a[4];
  #pragma unroll
  for (int kf = 0; kf < 4; ++kf) a[kf] = frag_gather_f32(arow + kf * 32, lane);
  const _Float16* w1h = w16 + W16_SCW1;
  #pragma unroll
  for (int j = 0; j < 4; ++j) {
    const _Float16* wrow = w1h + (size_t)(j * 16 + m) * 128;
    v8f c = {};
    #pragma unroll
    for (int kf = 0; kf < 4; ++kf) {
      const v16h bf = frag_gather_f16(wrow + kf * 32, lane);
      c = __builtin_amdgcn_wmma_f32_16x16x32_f16(false, a[kf], false, bf, (short)0, c, false, false);
    }
    const int col = j * 16 + m;
    const float bb = b1[col], wv = w2[col];
    #pragma unroll
    for (int i = 0; i < 8; ++i) {
      float t = c[i] + bb; t = t > 0.f ? t : 0.f;
      atomicAdd(&rowsum[wid][i + 8 * h], t * wv);
    }
  }
  __syncthreads();
  if (lane < 16) pscore[row0 + lane] = sigmf(rowsum[wid][lane] + b2[0]);
}

// ---------------- 9. path_weights softmax over K ----------------
__global__ void pw_kernel(const float* __restrict__ pscore, float* __restrict__ pw) {
  const int b = blockIdx.x;
  if (threadIdx.x == 0) {
    float mx = -1e30f;
    for (int k = 0; k < K_; ++k) mx = fmaxf(mx, pscore[b * K_ + k]);
    float e[K_], s = 0.f;
    for (int k = 0; k < K_; ++k) { e[k] = expf(pscore[b * K_ + k] - mx); s += e[k]; }
    for (int k = 0; k < K_; ++k) pw[b * K_ + k] = e[k] / s;
  }
}

extern "C" void kernel_launch(void* const* d_in, const int* in_sizes, int n_in,
                              void* d_out, int out_size, void* d_ws, size_t ws_size,
                              hipStream_t stream) {
  (void)in_sizes; (void)n_in; (void)out_size; (void)ws_size;
  const float* nf    = (const float*)d_in[0];
  const float* adj   = (const float*)d_in[1];
  const float* sp_w1 = (const float*)d_in[2];
  const float* sp_b1 = (const float*)d_in[3];
  const float* sp_w2 = (const float*)d_in[4];
  const float* sp_b2 = (const float*)d_in[5];
  const float* wih_f = (const float*)d_in[6];
  const float* whh_f = (const float*)d_in[7];
  const float* bih_f = (const float*)d_in[8];
  const float* bhh_f = (const float*)d_in[9];
  const float* wih_b = (const float*)d_in[10];
  const float* whh_b = (const float*)d_in[11];
  const float* bih_b = (const float*)d_in[12];
  const float* bhh_b = (const float*)d_in[13];
  const float* sc_w1 = (const float*)d_in[14];
  const float* sc_b1 = (const float*)d_in[15];
  const float* sc_w2 = (const float*)d_in[16];
  const float* sc_b2 = (const float*)d_in[17];

  float* out       = (float*)d_out;
  float* out_paths = out;                 // 32*16*16 = 8192
  float* out_pw    = out + 8192;          // 32*16    = 512
  float* out_pf    = out + 8192 + 512;    // 32*16*128 = 65536

  float*    fc     = (float*)d_ws;                 // 32768
  float*    score  = fc + 32768;                   // 32768
  float*    xw     = score + 32768;                // 2*8192*192 = 3145728
  float*    pscore = xw + 2 * (size_t)R_ * G_;     // 512
  int*      srcidx = (int*)(pscore + 512);         // 512
  int*      paths  = srcidx + 512;                 // 8192
  _Float16* x16    = (_Float16*)(paths + 8192);    // 524288 halves
  _Float16* w16    = x16 + 524288;                 // 61440 halves

  prep_kernel<<<(W16_TOTAL + 255) / 256, 256, 0, stream>>>(sp_w1, sc_w1, wih_f, wih_b,
                                                           whh_f, whh_b, w16);
  fc_kernel<<<B_ * N_, 64, 0, stream>>>(nf, fc);
  src_score_kernel<<<256, 256, 0, stream>>>(nf, w16, sp_b1, sp_w2, sp_b2, fc, score);
  topk_kernel<<<B_, 256, 0, stream>>>(score, srcidx);
  walk_kernel<<<B_ * K_, 256, 0, stream>>>(nf, adj, fc, srcidx, paths, out_paths);
  gather_kernel<<<2048, 256, 0, stream>>>(nf, paths, x16);
  xw_kernel<<<1536, 256, 0, stream>>>(x16, w16, xw);
  gru_scan_kernel<<<64, 128, 0, stream>>>(xw, w16, bih_f, bhh_f, bih_b, bhh_b, out_pf);
  pathscore_kernel<<<4, 256, 0, stream>>>(out_pf, w16, sc_b1, sc_w2, sc_b2, pscore);
  pw_kernel<<<B_, 32, 0, stream>>>(pscore, out_pw);
}
